// GNN_73280732004420
// MI455X (gfx1250) — matrix operand order repaired and checked by reading, not compiled, not run on hardware
//
#include <hip/hip_runtime.h>
#include <hip/hip_bf16.h>
#include <math.h>

typedef __attribute__((ext_vector_type(16))) _Float16 v16h;
typedef __attribute__((ext_vector_type(8)))  float    v8f;

#define N_DRUG 20000
#define N_PROT 30000
#define NN     (N_DRUG + N_PROT)   // 50000
#define F_DRUG 128
#define F_PROT 256
#define HCH    64
#define NE     1600000
#define NL     200000
#define LAYERS 3

// ---------------------------------------------------------------- utilities
__global__ void k_fill_f32(float* __restrict__ p, float v, int n) {
    int i = blockIdx.x * blockDim.x + threadIdx.x;
    if (i < n) p[i] = v;
}

// deg[dst] += 1 over all edges (deg pre-filled with 1.0 for the self loop)
__global__ void k_degree(const long long* __restrict__ dst_idx, float* __restrict__ deg) {
    int e = blockIdx.x * blockDim.x + threadIdx.x;
    if (e < NE) atomicAdd(&deg[(int)dst_idx[e]], 1.0f);
}

__global__ void k_rsqrt(const float* __restrict__ deg, float* __restrict__ dis, int n) {
    int i = blockIdx.x * blockDim.x + threadIdx.x;
    if (i < n) dis[i] = rsqrtf(deg[i]);
}

// ---------------------------------------------------------------- weight swizzle
// Pack W[K x 64] (f32) into the CDNA5 WMMA B-fragment lane layout (f16):
//   fragment element e of (k-chunk c, half-group g, column col) is K = c*32+8g+e (e<8), c*32+16+8g+(e-8) (e>=8)
//   stored contiguously so each lane loads its 16 halves as one 32B vector.
__global__ void k_swizzle_w(const float* __restrict__ W, _Float16* __restrict__ Bswz, int K) {
    int i = blockIdx.x * blockDim.x + threadIdx.x;
    if (i >= K * HCH) return;
    int kk  = i >> 6;          // K index
    int col = i & 63;          // column
    int c = kk >> 5, r = kk & 31, g = (r >> 3) & 1, e = (r & 7) | ((r >> 4) << 3);
    Bswz[((((size_t)c * HCH + col) * 2 + g) << 4) + e] = (_Float16)W[(size_t)kk * HCH + col];
}

// ---------------------------------------------------------------- WMMA GEMM
// C[M x 64] = op(A[M x K] (f32 -> f16 on the fly) @ W (pre-swizzled f16))
// 128 threads = 4 waves. Block b computes rows [32b, 32b+32) as two 16-row
// M-tiles sharing each wave's B fragment; wave w computes columns [16w,16w+16).
// K must be a multiple of 32. A-row indices of the (possibly absent) second
// tile are clamped so EXEC stays all-ones through the WMMAs (ISA 7.12 req);
// its store is guarded by a uniform branch.
__device__ __forceinline__ v16h load_a_frag(const float* __restrict__ ap) {
    float4 a0 = *(const float4*)(ap + 0);
    float4 a1 = *(const float4*)(ap + 4);
    float4 a2 = *(const float4*)(ap + 16);
    float4 a3 = *(const float4*)(ap + 20);
    v16h av;
    av[0]  = (_Float16)a0.x; av[1]  = (_Float16)a0.y;
    av[2]  = (_Float16)a0.z; av[3]  = (_Float16)a0.w;
    av[4]  = (_Float16)a1.x; av[5]  = (_Float16)a1.y;
    av[6]  = (_Float16)a1.z; av[7]  = (_Float16)a1.w;
    av[8]  = (_Float16)a2.x; av[9]  = (_Float16)a2.y;
    av[10] = (_Float16)a2.z; av[11] = (_Float16)a2.w;
    av[12] = (_Float16)a3.x; av[13] = (_Float16)a3.y;
    av[14] = (_Float16)a3.z; av[15] = (_Float16)a3.w;
    return av;
}

__global__ __launch_bounds__(128)
void k_gemm_wmma(const float* __restrict__ A, const _Float16* __restrict__ Bswz,
                 const float* __restrict__ bias, float* __restrict__ C,
                 int M, int K, int add_bias, int do_relu) {
    const int lane = threadIdx.x & 31;
    const int wav  = threadIdx.x >> 5;   // column tile 0..3
    const int g    = lane >> 4;          // half-wave group
    const int mr   = lane & 15;
    const int row0 = blockIdx.x * 32;
    const int col  = wav * 16 + mr;

    int ra1 = row0 + 16 + mr;
    if (ra1 > M - 1) ra1 = M - 1;        // clamp loads of absent tile
    const float* __restrict__ arow0 = A + (size_t)(row0 + mr) * (size_t)K;
    const float* __restrict__ arow1 = A + (size_t)ra1 * (size_t)K;

    v8f acc0 = {}, acc1 = {};
    for (int k0 = 0; k0 < K; k0 += 32) {
        // B fragment: one contiguous 32B vector per lane (pre-swizzled)
        const v16h bv = *(const v16h*)(Bswz + (((((size_t)(k0 >> 5)) * HCH + col) * 2 + g) << 4));
        v16h av0 = load_a_frag(arow0 + k0 + 8 * g);
        v16h av1 = load_a_frag(arow1 + k0 + 8 * g);
        acc0 = __builtin_amdgcn_wmma_f32_16x16x32_f16(false, av0, false, bv,
                                                      (short)0, acc0, false, false);
        acc1 = __builtin_amdgcn_wmma_f32_16x16x32_f16(false, av1, false, bv,
                                                      (short)0, acc1, false, false);
    }

    float b = add_bias ? bias[col] : 0.0f;
    #pragma unroll
    for (int r = 0; r < 8; ++r) {
        int m = row0 + r + 8 * g;
        float v = acc0[r] + b;
        if (do_relu) v = fmaxf(v, 0.0f);
        C[(size_t)m * HCH + col] = v;
    }
    if (row0 + 16 < M) {                 // uniform guard for second tile
        #pragma unroll
        for (int r = 0; r < 8; ++r) {
            int m = row0 + 16 + r + 8 * g;
            float v = acc1[r] + b;
            if (do_relu) v = fmaxf(v, 0.0f);
            C[(size_t)m * HCH + col] = v;
        }
    }
}

// ---------------------------------------------------------------- scatter
// 16 threads per edge, float4 per thread: agg[dst] += xw[src] * dis[s]*dis[d]
__global__ void k_scatter(const long long* __restrict__ src_idx,
                          const long long* __restrict__ dst_idx,
                          const float* __restrict__ dis,
                          const float* __restrict__ xw,
                          float* __restrict__ agg) {
    int t = blockIdx.x * blockDim.x + threadIdx.x;   // NE*16 = 25.6M threads
    if (t >= NE * 16) return;
    int e = t >> 4;
    int c = (t & 15) << 2;
    int s = (int)src_idx[e];
    int d = (int)dst_idx[e];
    float nrm = dis[s] * dis[d];
    float4 v = *(const float4*)(xw + (size_t)s * HCH + c);
    float* ap = agg + (size_t)d * HCH + c;
    atomicAdd(ap + 0, v.x * nrm);
    atomicAdd(ap + 1, v.y * nrm);
    atomicAdd(ap + 2, v.z * nrm);
    atomicAdd(ap + 3, v.w * nrm);
}

// x = relu(agg + xw * dis[n]^2 + bias[c])
__global__ void k_combine(const float* __restrict__ agg, const float* __restrict__ xw,
                          const float* __restrict__ dis, const float* __restrict__ bias,
                          float* __restrict__ x) {
    int i = blockIdx.x * blockDim.x + threadIdx.x;
    if (i >= NN * HCH) return;
    int n = i >> 6;
    int c = i & 63;
    float dn = dis[n];
    float v = agg[i] + xw[i] * dn * dn + bias[c];
    x[i] = fmaxf(v, 0.0f);
}

// sigmoid((x[s].lin_W+lb) * (x[d].lin_W+lb)) per label edge
__global__ void k_score(const float* __restrict__ x,
                        const long long* __restrict__ ls, const long long* __restrict__ ld,
                        const float* __restrict__ lw, const float* __restrict__ lb,
                        float* __restrict__ out) {
    int i = blockIdx.x * blockDim.x + threadIdx.x;
    if (i >= NL) return;
    const float* xs = x + (size_t)ls[i] * HCH;
    const float* xd = x + (size_t)ld[i] * HCH;
    float sd = lb[0], sp = lb[0];
    #pragma unroll 8
    for (int h = 0; h < HCH; ++h) {
        float w = lw[h];
        sd += xs[h] * w;
        sp += xd[h] * w;
    }
    out[i] = 1.0f / (1.0f + __expf(-sd * sp));
}

// ---------------------------------------------------------------- launcher
extern "C" void kernel_launch(void* const* d_in, const int* in_sizes, int n_in,
                              void* d_out, int out_size, void* d_ws, size_t ws_size,
                              hipStream_t stream) {
    const float*     x_drug = (const float*)d_in[0];
    const float*     x_prot = (const float*)d_in[1];
    const float*     W_drug = (const float*)d_in[2];
    const float*     b_drug = (const float*)d_in[3];
    const float*     W_prot = (const float*)d_in[4];
    const float*     b_prot = (const float*)d_in[5];
    const float*     conv_W = (const float*)d_in[6];
    const float*     conv_b = (const float*)d_in[7];
    const float*     lin_W  = (const float*)d_in[8];
    const float*     lin_b  = (const float*)d_in[9];
    const long long* ei     = (const long long*)d_in[10];  // [2, NE]
    const long long* els    = (const long long*)d_in[11];  // [NL]
    const long long* eld    = (const long long*)d_in[12];  // [NL]
    float* out = (float*)d_out;

    // workspace layout (256B-aligned slices)
    char* ws = (char*)d_ws;
    size_t off = 0;
    auto take = [&](size_t bytes) -> char* {
        char* p = ws + off;
        off = (off + bytes + 255) & ~(size_t)255;
        return p;
    };
    float*    deg  = (float*)take((size_t)NN * 4);
    float*    dis  = (float*)take((size_t)NN * 4);
    float*    x    = (float*)take((size_t)NN * HCH * 4);
    float*    xw   = (float*)take((size_t)NN * HCH * 4);
    float*    agg  = (float*)take((size_t)NN * HCH * 4);
    _Float16* wd16 = (_Float16*)take((size_t)F_DRUG * HCH * 2);
    _Float16* wp16 = (_Float16*)take((size_t)F_PROT * HCH * 2);
    _Float16* wc16 = (_Float16*)take((size_t)LAYERS * HCH * HCH * 2);

    const long long* src = ei;
    const long long* dst = ei + NE;

    // 1) degrees + normalization
    k_fill_f32<<<(NN + 255) / 256, 256, 0, stream>>>(deg, 1.0f, NN);
    k_degree<<<(NE + 255) / 256, 256, 0, stream>>>(dst, deg);
    k_rsqrt<<<(NN + 255) / 256, 256, 0, stream>>>(deg, dis, NN);

    // 2) weights -> f16, pre-swizzled into WMMA B-fragment layout
    k_swizzle_w<<<(F_DRUG * HCH + 255) / 256, 256, 0, stream>>>(W_drug, wd16, F_DRUG);
    k_swizzle_w<<<(F_PROT * HCH + 255) / 256, 256, 0, stream>>>(W_prot, wp16, F_PROT);
    for (int l = 0; l < LAYERS; ++l)
        k_swizzle_w<<<(HCH * HCH + 255) / 256, 256, 0, stream>>>(
            conv_W + (size_t)l * HCH * HCH, wc16 + (size_t)l * HCH * HCH, HCH);

    // 3) typed input projections (WMMA, +bias, +relu) into x
    k_gemm_wmma<<<(N_DRUG + 31) / 32, 128, 0, stream>>>(x_drug, wd16, b_drug, x,
                                                        N_DRUG, F_DRUG, 1, 1);
    k_gemm_wmma<<<(N_PROT + 31) / 32, 128, 0, stream>>>(x_prot, wp16, b_prot,
                                                        x + (size_t)N_DRUG * HCH,
                                                        N_PROT, F_PROT, 1, 1);

    // 4) GCN layers
    for (int l = 0; l < LAYERS; ++l) {
        k_gemm_wmma<<<(NN + 31) / 32, 128, 0, stream>>>(x, wc16 + (size_t)l * HCH * HCH,
                                                        nullptr, xw, NN, HCH, 0, 0);
        k_fill_f32<<<(NN * HCH + 255) / 256, 256, 0, stream>>>(agg, 0.0f, NN * HCH);
        k_scatter<<<(NE * 16 + 255) / 256, 256, 0, stream>>>(src, dst, dis, xw, agg);
        k_combine<<<(NN * HCH + 255) / 256, 256, 0, stream>>>(agg, xw, dis,
                                                              conv_b + (size_t)l * HCH, x);
    }

    // 5) edge scoring head
    k_score<<<(NL + 255) / 256, 256, 0, stream>>>(x, els, eld, lin_W, lin_b, out);
    (void)in_sizes; (void)n_in; (void)out_size; (void)ws_size;
}
